// CustomBidirectionalRNN_76063870812448
// MI455X (gfx1250) — matrix operand order repaired and checked by reading, not compiled, not run on hardware
//
#include <hip/hip_runtime.h>
#include <cmath>

typedef __bf16 bf16;
typedef __attribute__((ext_vector_type(8)))  bf16  bf16x8;
typedef __attribute__((ext_vector_type(16))) bf16  bf16x16;
typedef __attribute__((ext_vector_type(8)))  float floatx8;

#define B_ 32
#define S_ 1024
#define I_ 1024
#define H_ 512
#define L_ 2

// ---------------------------------------------------------------------------
// CDNA5 async global->LDS copy (ASYNCcnt-tracked, bypasses VGPRs)
// ---------------------------------------------------------------------------
__device__ __forceinline__ void async_cp_b128(const bf16* gptr, bf16* lptr) {
  unsigned loff = (unsigned)(uintptr_t)lptr;       // LDS addr = flat[31:0]
  asm volatile("global_load_async_to_lds_b128 %0, %1, off"
               :: "v"(loff), "v"(gptr) : "memory");
}
__device__ __forceinline__ void wait_async0() {
  asm volatile("s_wait_asynccnt 0" ::: "memory");
}

// ---------------------------------------------------------------------------
// Converters
// ---------------------------------------------------------------------------
// X (B,S,I) f32  ->  Xbf (S,B,I) bf16   (GEMM row = s*B + b, K-contiguous)
__global__ void conv_x_kernel(const float* __restrict__ X, bf16* __restrict__ Xbf) {
  size_t idx = (size_t)blockIdx.x * blockDim.x + threadIdx.x;   // < B*S*I = 2^25
  int i = (int)(idx & (I_ - 1));
  int s = (int)((idx >> 10) & (S_ - 1));
  int b = (int)(idx >> 20);
  Xbf[((size_t)s * B_ + b) * I_ + i] = (bf16)X[idx];
}

// src (L, chunk) f32 -> dst[(l*2 + d)*chunk + r] bf16   (pack fwd/rev interleaved)
__global__ void conv_w_kernel(const float* __restrict__ src, bf16* __restrict__ dst,
                              int chunk, int d, int n) {
  int idx = blockIdx.x * blockDim.x + threadIdx.x;
  if (idx >= n) return;
  int l = idx / chunk, r = idx - l * chunk;
  dst[(size_t)(l * 2 + d) * chunk + r] = (bf16)src[idx];
}

// ---------------------------------------------------------------------------
// Phase 1: P[d] = A @ W[d]^T + bias[d]     (A: [M][K] bf16, W: [N][K] bf16)
// Block tile 128x128, K-step 32, 8 waves (2x4), wave tile 64x32 = 8 WMMA tiles.
// Async double-buffered LDS staging: one barrier + one s_wait_asynccnt per step.
// ---------------------------------------------------------------------------
#define BLK_M 128
#define BLK_N 128
#define BLK_K 32
#define LDSTR 40   // LDS row stride (elements), padded vs 32 to de-conflict banks

__global__ __launch_bounds__(256) void gemm_bf16_kernel(
    const bf16* __restrict__ A,       // [M][K]
    const bf16* __restrict__ Wall,    // [2][N][K]  (dir-major)
    const float* __restrict__ biasF,
    const float* __restrict__ biasR,
    float* __restrict__ Pall,         // [2][M][N]
    int M, int N, int K)
{
  __shared__ bf16 As[2][BLK_M * LDSTR];
  __shared__ bf16 Bs[2][BLK_N * LDSTR];

  const int tid  = threadIdx.x;
  const int wave = tid >> 5;
  const int lane = tid & 31;
  const int half = lane >> 4;
  const int lm   = lane & 15;
  const int mw   = wave >> 2;            // 0..1
  const int nw   = wave & 3;             // 0..3

  const int d  = blockIdx.z;
  const int m0 = blockIdx.y * BLK_M;
  const int n0 = blockIdx.x * BLK_N;

  const bf16*  Wd   = Wall + (size_t)d * N * K;
  const float* bias = d ? biasR : biasF;
  float*       Pout = Pall + (size_t)d * M * N;

  floatx8 zero = {0.f, 0.f, 0.f, 0.f, 0.f, 0.f, 0.f, 0.f};
  floatx8 acc[4][2];
#pragma unroll
  for (int mi = 0; mi < 4; ++mi)
#pragma unroll
    for (int ni = 0; ni < 2; ++ni) acc[mi][ni] = zero;

  const int lrow = tid >> 2;             // 0..63
  const int lseg = (tid & 3) * 8;        // 0,8,16,24

  // Per-thread DMA job: 4x16B (two A rows, two W rows) per tile.
  auto issue_tile = [&](int k0, int buf) {
    async_cp_b128(&A [(size_t)(m0 + lrow)      * K + k0 + lseg], &As[buf][ lrow       * LDSTR + lseg]);
    async_cp_b128(&A [(size_t)(m0 + lrow + 64) * K + k0 + lseg], &As[buf][(lrow + 64) * LDSTR + lseg]);
    async_cp_b128(&Wd[(size_t)(n0 + lrow)      * K + k0 + lseg], &Bs[buf][ lrow       * LDSTR + lseg]);
    async_cp_b128(&Wd[(size_t)(n0 + lrow + 64) * K + k0 + lseg], &Bs[buf][(lrow + 64) * LDSTR + lseg]);
  };

  const int nk = K / BLK_K;
  issue_tile(0, 0);

  for (int ki = 0; ki < nk; ++ki) {
    const int buf = ki & 1;
    wait_async0();                       // my DMA into buf done
    __syncthreads();                     // everyone's DMA done + prior reads done
    if (ki + 1 < nk) issue_tile((ki + 1) * BLK_K, buf ^ 1);

    union { bf16x16 v; bf16x8 h[2]; } afr[4], bfr[2];
#pragma unroll
    for (int ni = 0; ni < 2; ++ni) {
      int n = nw * 32 + ni * 16 + lm;    // lane = column; K contiguous per lane
      bfr[ni].h[0] = *(const bf16x8*)&Bs[buf][n * LDSTR + 16 * half];
      bfr[ni].h[1] = *(const bf16x8*)&Bs[buf][n * LDSTR + 16 * half + 8];
    }
#pragma unroll
    for (int mi = 0; mi < 4; ++mi) {
      int m = mw * 64 + mi * 16 + lm;    // lane = row; two 8-elt K runs
      afr[mi].h[0] = *(const bf16x8*)&As[buf][m * LDSTR + 16 * half];
      afr[mi].h[1] = *(const bf16x8*)&As[buf][m * LDSTR + 16 * half + 8];
    }
#pragma unroll
    for (int mi = 0; mi < 4; ++mi)
#pragma unroll
      for (int ni = 0; ni < 2; ++ni)
        acc[mi][ni] = __builtin_amdgcn_wmma_f32_16x16x32_bf16(
            false, afr[mi].v, false, bfr[ni].v, (short)0, acc[mi][ni],
            false, false);
  }

#pragma unroll
  for (int mi = 0; mi < 4; ++mi) {
    int rowb = m0 + mw * 64 + mi * 16 + 8 * half;   // D: M = v + 8*half
#pragma unroll
    for (int ni = 0; ni < 2; ++ni) {
      int col = n0 + nw * 32 + ni * 16 + lm;        // D: N = lane%16
      float bv = bias[col];
#pragma unroll
      for (int v = 0; v < 8; ++v)
        Pout[(size_t)(rowb + v) * N + col] = acc[mi][ni][v] + bv;
    }
  }
}

// ---------------------------------------------------------------------------
// Phase 2: sequential scan  h <- tanh(P[s] + h @ Whh^T)
// grid = (2 dirs, 2 batch-tiles of 16) ; block = 1024 thr = 32 waves.
// Wave w owns output columns [16w,16w+16); its Whh slice (512x16 bf16) is held
// in 16 v16bf fragments (128 VGPRs) for the whole scan. h is double-buffered
// in LDS -> one barrier per step; K-loop uses two accumulators to shorten the
// WMMA D->C dependency chain.
// ---------------------------------------------------------------------------
__global__ __launch_bounds__(1024) void rnn_scan_kernel(
    const float* __restrict__ Pall,    // [2][S][B][H]
    const bf16*  __restrict__ Whhall,  // [2][H][H]  (this layer, dir-major)
    bf16*  __restrict__ Y0,            // layer0: [S][B][2H] bf16 out (else unused)
    float* __restrict__ outp,          // layer1: [B][S][2H] f32 out (else unused)
    float* __restrict__ hid,           // [2L][B][H] final hidden
    int layer)
{
  __shared__ bf16 hsh[2][16 * H_];     // double-buffered h: 16 rows x H each

  const int d    = blockIdx.x;
  const int mt   = blockIdx.y;         // batch rows [16*mt, 16*mt+16)
  const int tid  = threadIdx.x;
  const int wave = tid >> 5;           // n-tile id 0..31
  const int lane = tid & 31;
  const int half = lane >> 4;
  const int lm   = lane & 15;

  const float* P   = Pall + (size_t)d * S_ * B_ * H_;
  const bf16*  Whh = Whhall + (size_t)d * H_ * H_;

  // Pin this wave's B-operand (Whh columns) in VGPRs: 16 chunks x v16bf.
  union BF { bf16x16 v; bf16x8 h[2]; };
  BF bfr[16];
  {
    const bf16* wrow = Whh + (size_t)(wave * 16 + lm) * H_;   // [N][K] row
#pragma unroll
    for (int c = 0; c < 16; ++c) {
      bfr[c].h[0] = *(const bf16x8*)&wrow[c * 32 + 8 * half];
      bfr[c].h[1] = *(const bf16x8*)&wrow[c * 32 + 16 + 8 * half];
    }
  }

  for (int i = tid; i < 16 * H_; i += 1024) hsh[0][i] = (bf16)0.f;  // h0 = 0
  __syncthreads();

  const int col = wave * 16 + lm;      // output column (D: N = lane%16)
  floatx8 zero = {0.f, 0.f, 0.f, 0.f, 0.f, 0.f, 0.f, 0.f};

  for (int t = 0; t < S_; ++t) {
    const int s   = d ? (S_ - 1 - t) : t;
    const int cur = t & 1, nxt = cur ^ 1;

    if (t + 1 < S_) {                  // hide next step's P fetch (L2/HBM)
      const int sn = d ? (S_ - 2 - t) : (t + 1);
      __builtin_prefetch(&P[((size_t)sn * B_ + mt * 16 + lm) * H_ + wave * 16], 0, 1);
    }

    floatx8 acc0 = zero, acc1 = zero;  // split D->C chain across two accs
#pragma unroll
    for (int c = 0; c < 16; c += 2) {
      BF afr0, afr1;
      const bf16* hrow = &hsh[cur][lm * H_ + c * 32];
      afr0.h[0] = *(const bf16x8*)&hrow[8 * half];
      afr0.h[1] = *(const bf16x8*)&hrow[16 + 8 * half];
      afr1.h[0] = *(const bf16x8*)&hrow[32 + 8 * half];
      afr1.h[1] = *(const bf16x8*)&hrow[48 + 8 * half];
      acc0 = __builtin_amdgcn_wmma_f32_16x16x32_bf16(
          false, afr0.v, false, bfr[c].v,     (short)0, acc0, false, false);
      acc1 = __builtin_amdgcn_wmma_f32_16x16x32_bf16(
          false, afr1.v, false, bfr[c + 1].v, (short)0, acc1, false, false);
    }

    float hv[8];
#pragma unroll
    for (int v = 0; v < 8; ++v) {
      int lr    = v + 8 * half;        // local batch row (D layout)
      int batch = mt * 16 + lr;
      float x = acc0[v] + acc1[v] + P[((size_t)s * B_ + batch) * H_ + col];
      hv[v] = tanhf(x);
      hsh[nxt][lr * H_ + col] = (bf16)hv[v];   // write next buffer (no race)
    }
    if (layer == 0) {
#pragma unroll
      for (int v = 0; v < 8; ++v) {
        int batch = mt * 16 + v + 8 * half;
        Y0[((size_t)s * B_ + batch) * (2 * H_) + d * H_ + col] = (bf16)hv[v];
      }
    } else {
#pragma unroll
      for (int v = 0; v < 8; ++v) {
        int batch = mt * 16 + v + 8 * half;
        outp[((size_t)batch * S_ + s) * (2 * H_) + d * H_ + col] = hv[v];
      }
    }
    if (t == S_ - 1) {
#pragma unroll
      for (int v = 0; v < 8; ++v) {
        int batch = mt * 16 + v + 8 * half;
        hid[((size_t)(2 * layer + d) * B_ + batch) * H_ + col] = hv[v];
      }
    }
    __syncthreads();   // next-h visible; also fences reads of cur before reuse
  }
}

// ---------------------------------------------------------------------------
// Host launcher
// ---------------------------------------------------------------------------
extern "C" void kernel_launch(void* const* d_in, const int* in_sizes, int n_in,
                              void* d_out, int out_size, void* d_ws, size_t ws_size,
                              hipStream_t stream) {
  const float* X     = (const float*)d_in[0];
  const float* Wih   = (const float*)d_in[1];
  const float* Whh   = (const float*)d_in[2];
  const float* biasF = (const float*)d_in[3];
  const float* WihR  = (const float*)d_in[4];
  const float* WhhR  = (const float*)d_in[5];
  const float* biasR = (const float*)d_in[6];

  const size_t M = (size_t)S_ * B_;    // 32768 GEMM rows

  char* ws = (char*)d_ws;
  bf16*  Xbf  = (bf16*)ws;                                   // M*I bf16   (64 MB)
  bf16*  Y0   = (bf16*)(ws + M * I_ * 2);                    // M*2H bf16  (64 MB)
  float* Pbuf = (float*)(ws + M * I_ * 2 + M * 2 * H_ * 2);  // 2*M*H f32 (128 MB)
  bf16*  WihB = (bf16*)((char*)Pbuf + (size_t)2 * M * H_ * 4);        // [L][2][H][I]
  bf16*  WhhB = (bf16*)((char*)WihB + (size_t)L_ * 2 * H_ * I_ * 2);  // [L][2][H][H]

  float* outp = (float*)d_out;                       // (B,S,2H)
  float* hid  = outp + (size_t)B_ * S_ * 2 * H_;     // (2L,B,H)

  // bf16 conversions / transposes
  conv_x_kernel<<<(B_ * S_ * I_) / 256, 256, 0, stream>>>(X, Xbf);
  const int nIh = L_ * H_ * I_, nHh = L_ * H_ * H_;
  conv_w_kernel<<<(nIh + 255) / 256, 256, 0, stream>>>(Wih,  WihB, H_ * I_, 0, nIh);
  conv_w_kernel<<<(nIh + 255) / 256, 256, 0, stream>>>(WihR, WihB, H_ * I_, 1, nIh);
  conv_w_kernel<<<(nHh + 255) / 256, 256, 0, stream>>>(Whh,  WhhB, H_ * H_, 0, nHh);
  conv_w_kernel<<<(nHh + 255) / 256, 256, 0, stream>>>(WhhR, WhhB, H_ * H_, 1, nHh);

  dim3 ggrid(H_ / BLK_N, (unsigned)(M / BLK_M), 2);  // (4, 256, 2)

  // Layer 0: P = X @ Wih^T + b ; scan ; Y0 = concat(fwd, rev) in bf16
  gemm_bf16_kernel<<<ggrid, 256, 0, stream>>>(
      Xbf, WihB, biasF, biasR, Pbuf, (int)M, H_, I_);
  rnn_scan_kernel<<<dim3(2, 2), 1024, 0, stream>>>(
      Pbuf, WhhB, Y0, nullptr, hid, 0);

  // Layer 1: P = Y0 @ Wih[1]^T + b[1] ; scan ; write final_output f32
  gemm_bf16_kernel<<<ggrid, 256, 0, stream>>>(
      Y0, WihB + (size_t)2 * H_ * I_, biasF + H_, biasR + H_, Pbuf,
      (int)M, H_, 2 * H_);
  rnn_scan_kernel<<<dim3(2, 2), 1024, 0, stream>>>(
      Pbuf, WhhB + (size_t)2 * H_ * H_, nullptr, outp, hid, 1);
}
